// EchoStateNetwork_31662498906435
// MI455X (gfx1250) — matrix operand order repaired and checked by reading, not compiled, not run on hardware
//
#include <hip/hip_runtime.h>
#include <hip/hip_bf16.h>

// ---------------------------------------------------------------------------
// Echo State Network for MI455X (gfx1250, wave32, WMMA bf16)
//   B=32, C=4, T=512, I=64, R=1024, LEAK=0.5
//   out[b,c,t,r] : [32,4,512,1024] f32
// ---------------------------------------------------------------------------

typedef __attribute__((ext_vector_type(16))) __bf16        v16bf;
typedef __attribute__((ext_vector_type(8)))  float         v8f;
typedef __attribute__((ext_vector_type(4)))  unsigned int  v4u;

union FragU {
    v16bf v;
    v4u   q[2];
};

// round-to-nearest-even f32 -> bf16 (bit pattern)
__device__ __forceinline__ unsigned short f2bf(float f) {
    unsigned u = __builtin_bit_cast(unsigned, f);
    u += 0x7FFFu + ((u >> 16) & 1u);
    return (unsigned short)(u >> 16);
}

// Load one 16x32 bf16 A-fragment for lane (row fixed, K contiguous):
//   elements e0..e7  at  k0+half*8   .. +7   (16 bytes)
//   elements e8..e15 at  k0+16+half*8 .. +7  (16 bytes)
// caller passes p = rowBase + k0 + half*8 (16B aligned)
__device__ __forceinline__ v16bf load_frag16(const unsigned short* p) {
    FragU f;
    f.q[0] = *(const v4u*)(p);
    f.q[1] = *(const v4u*)(p + 16);   // +16 elements = +32 bytes
    return f.v;
}

// ---------------------------------------------------------------------------
// Kernel 1: repack x[B,C,T,I] (f32) -> Xbf[T*B, C*I] (bf16), i.e. X2[t,b,k]
// ---------------------------------------------------------------------------
__global__ void esn_repack_x(const float* __restrict__ x,
                             unsigned short* __restrict__ xbf) {
    const int tb = blockIdx.x;          // t*32 + b
    const int t  = tb >> 5;
    const int b  = tb & 31;
    const int j  = threadIdx.x;         // 0..255 == c*64 + i
    const int c  = j >> 6;
    const int i  = j & 63;
    const float v = x[(((b << 2) + c) * 512 + t) * 64 + i];
    xbf[tb * 256 + j] = f2bf(v);
}

// ---------------------------------------------------------------------------
// Kernel 2: persistent scan. 64 WGs x 128 threads. WG g owns columns
// [16g, 16g+16). Wave w owns rows [32w, 32w+32) (m = b*C + c, M=128 total).
// ---------------------------------------------------------------------------
__global__ __launch_bounds__(128)
void esn_scan(const float* __restrict__ Wi,          // [1024][256] f32
              const float* __restrict__ W,           // [1024][1024] f32
              const unsigned short* __restrict__ Xbf,// [512*32][256] bf16
              unsigned short* __restrict__ Hb0,      // [128][1024] bf16 (t=0 zeroed)
              unsigned short* __restrict__ Hb1,      // [128][1024] bf16
              unsigned int* __restrict__ bar,        // zeroed monotonic counter
              float* __restrict__ out)               // [(m*512 + t)*1024 + n]
{
    // Pre-swizzled B fragments: [ktile][lane][e] bf16, 32B per lane per ktile.
    __shared__ __align__(32) unsigned short ldsW [32 * 32 * 16];  // 32 KB
    __shared__ __align__(32) unsigned short ldsWi[ 8 * 32 * 16];  //  8 KB

    const int n0  = blockIdx.x << 4;
    const int tid = threadIdx.x;

    // Fill W slice in B-fragment layout: lane l holds col n=n0+(l&15),
    // element e -> k = kt*32 + (e&7) + (l>>4)*8 + (e>>3)*16.  B[k,n] = W[k*1024+n].
    for (int idx = tid; idx < 32 * 32 * 16; idx += 128) {
        const int kt = idx >> 9;
        const int ln = (idx >> 4) & 31;
        const int e  = idx & 15;
        const int k  = kt * 32 + (e & 7) + ((ln >> 4) << 3) + ((e >> 3) << 4);
        const int n  = n0 + (ln & 15);
        ldsW[idx] = f2bf(W[k * 1024 + n]);
    }
    // Wi slice: B'[k,n] = Wi[n*256 + k]  (u = X2 @ Wi^T)
    for (int idx = tid; idx < 8 * 32 * 16; idx += 128) {
        const int kt = idx >> 9;
        const int ln = (idx >> 4) & 31;
        const int e  = idx & 15;
        const int k  = kt * 32 + (e & 7) + ((ln >> 4) << 3) + ((e >> 3) << 4);
        const int n  = n0 + (ln & 15);
        ldsWi[idx] = f2bf(Wi[n * 256 + k]);
    }
    __syncthreads();

    const int wv   = tid >> 5;
    const int lane = tid & 31;
    const int hh   = lane >> 4;        // half
    const int lr   = lane & 15;
    const int m0   = wv << 5;          // 32 rows per wave (2 M-tiles)
    const int kofs = hh << 3;          // A-fragment k offset for this half

    v8f h0, h1, acc0, acc1;
    #pragma unroll
    for (int v = 0; v < 8; ++v) { h0[v] = 0.0f; h1[v] = 0.0f; }

    const unsigned short* Hc = Hb0;
    unsigned short*       Hn = Hb1;

    for (int t = 0; t < 512; ++t) {
        #pragma unroll
        for (int v = 0; v < 8; ++v) { acc0[v] = 0.0f; acc1[v] = 0.0f; }

        // ---- input projection: A rows are b = m>>2 (broadcast over channel)
        const unsigned short* Xt  = Xbf + (t << 5) * 256;
        const unsigned short* xr0 = Xt + ((m0 + lr) >> 2) * 256;
        const unsigned short* xr1 = Xt + ((m0 + 16 + lr) >> 2) * 256;
        #pragma unroll
        for (int kt = 0; kt < 8; ++kt) {
            const v16bf bf = *(const v16bf*)(&ldsWi[(kt * 32 + lane) * 16]);
            const v16bf a0 = load_frag16(xr0 + kt * 32 + kofs);
            const v16bf a1 = load_frag16(xr1 + kt * 32 + kofs);
            acc0 = __builtin_amdgcn_wmma_f32_16x16x32_bf16(
                       false, a0, false, bf, (short)0, acc0, false, false);
            acc1 = __builtin_amdgcn_wmma_f32_16x16x32_bf16(
                       false, a1, false, bf, (short)0, acc1, false, false);
        }

        // ---- recurrent term: h @ W  (H ping-pong buffer, L2-resident)
        const unsigned short* hr0 = Hc + (m0 + lr) * 1024;
        const unsigned short* hr1 = Hc + (m0 + 16 + lr) * 1024;
        #pragma unroll 4
        for (int kt = 0; kt < 32; ++kt) {
            const v16bf bf = *(const v16bf*)(&ldsW[(kt * 32 + lane) * 16]);
            const v16bf a0 = load_frag16(hr0 + kt * 32 + kofs);
            const v16bf a1 = load_frag16(hr1 + kt * 32 + kofs);
            acc0 = __builtin_amdgcn_wmma_f32_16x16x32_bf16(
                       false, a0, false, bf, (short)0, acc0, false, false);
            acc1 = __builtin_amdgcn_wmma_f32_16x16x32_bf16(
                       false, a1, false, bf, (short)0, acc1, false, false);
        }

        // ---- leaky update + stores (D layout: VGPR v -> row m0 + v + half*8)
        const int col = n0 + lr;
        #pragma unroll
        for (int v = 0; v < 8; ++v) {
            const int r0 = m0 + v + (hh << 3);
            const float hn0 = 0.5f * tanhf(acc0[v]) + 0.5f * h0[v];
            h0[v] = hn0;
            Hn[r0 * 1024 + col] = f2bf(hn0);
            out[(r0 * 512 + t) * 1024 + col] = hn0;

            const int r1 = m0 + 16 + v + (hh << 3);
            const float hn1 = 0.5f * tanhf(acc1[v]) + 0.5f * h1[v];
            h1[v] = hn1;
            Hn[r1 * 1024 + col] = f2bf(hn1);
            out[(r1 * 512 + t) * 1024 + col] = hn1;
        }

        // ---- device-wide barrier (monotonic counter, target (t+1)*64)
        __threadfence();
        __syncthreads();
        if (tid == 0) {
            __hip_atomic_fetch_add(bar, 1u, __ATOMIC_ACQ_REL,
                                   __HIP_MEMORY_SCOPE_AGENT);
            const unsigned target = (unsigned)(t + 1) * 64u;
            while (__hip_atomic_load(bar, __ATOMIC_ACQUIRE,
                                     __HIP_MEMORY_SCOPE_AGENT) < target) {
                __builtin_amdgcn_s_sleep(1);
            }
        }
        __syncthreads();

        // ping-pong
        const unsigned short* tmp = Hc;
        Hc = Hn;
        Hn = (unsigned short*)tmp;
    }
}

// ---------------------------------------------------------------------------
// Workspace layout (zeroed region first):
//   [0,        262144)  Hb0  (bf16, must be 0 at t=0)
//   [262144,   524288)  Hb1
//   [524288,   524544)  barrier counter (+pad)
//   [524544, 8913152)   Xbf  (bf16 repacked input)
// ---------------------------------------------------------------------------
extern "C" void kernel_launch(void* const* d_in, const int* in_sizes, int n_in,
                              void* d_out, int out_size, void* d_ws, size_t ws_size,
                              hipStream_t stream) {
    const float* x  = (const float*)d_in[0];   // [32,4,512,64]
    const float* Wi = (const float*)d_in[1];   // [1024,256]
    const float* W  = (const float*)d_in[2];   // [1024,1024]
    float* out      = (float*)d_out;           // [32,4,512,1024]

    char* ws = (char*)d_ws;
    unsigned short* Hb0 = (unsigned short*)(ws);
    unsigned short* Hb1 = (unsigned short*)(ws + 262144);
    unsigned int*   bar = (unsigned int*)(ws + 524288);
    unsigned short* Xbf = (unsigned short*)(ws + 524544);

    (void)in_sizes; (void)n_in; (void)out_size; (void)ws_size;

    (void)hipMemsetAsync(ws, 0, 524544, stream);  // zero H ping-pong + barrier

    esn_repack_x<<<dim3(512 * 32), dim3(256), 0, stream>>>(x, Xbf);
    esn_scan<<<dim3(64), dim3(128), 0, stream>>>(Wi, W, Xbf, Hb0, Hb1, bar, out);
}
